// GNN2_50483045597217
// MI455X (gfx1250) — matrix operand order repaired and checked by reading, not compiled, not run on hardware
//
#include <hip/hip_runtime.h>
#include <hip/hip_bf16.h>
#include <math.h>

typedef __attribute__((ext_vector_type(16))) _Float16     v16h;
typedef __attribute__((ext_vector_type(8)))  float        v8f;
typedef __attribute__((ext_vector_type(4)))  float        v4f;
typedef __attribute__((ext_vector_type(4)))  unsigned int v4u;
typedef __attribute__((ext_vector_type(4)))  int          v4i;
typedef __attribute__((ext_vector_type(8)))  int          v8i;

#define TPAST 12
#define HIDC  32
#define TEMPC 76

__device__ __forceinline__ float lrelu(float v, float s) { return v > 0.0f ? v : v * s; }

// order-preserving float -> uint key (for atomicMax on floats incl. negatives)
__device__ __forceinline__ unsigned fkey(float f) {
    unsigned u = __float_as_uint(f);
    return (u & 0x80000000u) ? ~u : (u | 0x80000000u);
}
__device__ __forceinline__ float funkey(unsigned k) {
    return __uint_as_float((k & 0x80000000u) ? (k & 0x7fffffffu) : ~k);
}

// ---------------------------------------------------------------------------
// Stage `bytes` of pre-packed B fragments (global) into LDS.
// Preferred path: Tensor Data Mover (TDM). D# group0/group1 built per
// CDNA5 ISA: count=1, type=2("image"), data_size=4B,
// one row of bytes/4 dwords (tile_dim0=tensor_dim0=dwords, tile_dim1=1).
// This toolchain's builtin is the 6-arg form:
//   (uint32x4 g0, int32x8 g1, int32x4 g2, int32x4 g3, int32x8 pad, i32 cpol)
// Completion via TENSORcnt, then workgroup barrier publishes LDS.
// ---------------------------------------------------------------------------
__device__ __forceinline__ void stage_B(const _Float16* __restrict__ gB,
                                        _Float16* sB, unsigned bytes) {
#if defined(__AMDGCN__) && __has_builtin(__builtin_amdgcn_tensor_load_to_lds) && __has_builtin(__builtin_amdgcn_s_wait_tensorcnt)
    if ((threadIdx.x >> 5) == 0) {               // one TDM op per block (wave 0)
        unsigned long long ga = (unsigned long long)(size_t)gB;
        unsigned lds = (unsigned)(size_t)sB;     // LDS aperture: low 32 bits = LDS byte offset
        unsigned dw  = bytes >> 2;               // dwords to move
        v4u g0;
        g0.x = 1u;                                        // count=1 (user descriptor)
        g0.y = lds;                                       // lds_addr
        g0.z = (unsigned)ga;                              // global_addr[31:0]
        g0.w = (unsigned)(ga >> 32) | (2u << 30);         // global_addr[56:32] | type=2
        v8i g1;
        g1[0] = (int)(2u << 16);                          // data_size=2 (4 bytes)
        g1[1] = (int)((dw & 0xFFFFu) << 16);              // tensor_dim0[15:0]
        g1[2] = (int)(((dw >> 16) & 0xFFFFu) | (1u << 16)); // tensor_dim0[31:16] | tensor_dim1=1
        g1[3] = (int)((dw & 0xFFFFu) << 16);              // tile_dim0 = dwords
        g1[4] = 1;                                        // tile_dim1 = 1
        g1[5] = (int)dw;                                  // tensor_dim0_stride[31:0]
        g1[6] = 0;
        g1[7] = 0;
        v4i zz4 = {0, 0, 0, 0};
        v8i zz8 = {0, 0, 0, 0, 0, 0, 0, 0};
        __builtin_amdgcn_tensor_load_to_lds(g0, g1, zz4, zz4, zz8, 0);
        __builtin_amdgcn_s_wait_tensorcnt(0);
    }
#else
    {   // cooperative fallback copy (also taken by the host compilation pass)
        const v4u* s = (const v4u*)gB;
        v4u* d = (v4u*)sB;
        unsigned n16 = bytes >> 4;
        for (unsigned i = threadIdx.x; i < n16; i += blockDim.x) d[i] = s[i];
    }
#endif
    __syncthreads();
}

// ---------------------------------------------------------------------------
// Weight packing: W[K,Cout] (f32, row-major) -> per-fragment f16 B layout.
// B fragment (16x16x32 f16 WMMA): lane l holds column n = l&15; half j holds
// k_local = (l>>4)*8 + j (j<8), 16 + (l>>4)*8 + (j-8) (j>=8). Fragments stored [kt*NT+nt][lane][j] contiguous.
// ---------------------------------------------------------------------------
__global__ void pack_b(const float* __restrict__ W, _Float16* __restrict__ Bp,
                       int K, int Cout, int KT, int NT) {
    int t = blockIdx.x * blockDim.x + threadIdx.x;
    int total = KT * NT * 32;
    if (t >= total) return;
    int lane = t & 31;
    int frag = t >> 5;
    int nt = frag % NT;
    int kt = frag / NT;
    int n  = nt * 16 + (lane & 15);
    int g  = lane >> 4;
    _Float16* o = Bp + (size_t)frag * 32 * 16 + (size_t)lane * 16;
    for (int j = 0; j < 16; ++j) {
        int k = kt * 32 + (j < 8 ? g * 8 + j : 16 + g * 8 + (j - 8));
        float v = (k < K && n < Cout) ? W[(size_t)k * Cout + n] : 0.0f;
        o[j] = (_Float16)v;
    }
}

// ---------------------------------------------------------------------------
// WMMA GEMM: C[nrows, CCOLS] = A[nrows, ACOLS] * B  (f16 in, f32 accum)
// 8 waves/block, one 16-row strip per wave across NT column tiles, KT k-steps.
// A loads: 4x float4 per k-step (two contiguous 8-float runs per lane),
// compile-time maskable via ACOLS. B read from LDS (TDM-staged). C written
// through a per-wave LDS tile and streamed out as aligned b128 stores.
// ---------------------------------------------------------------------------
template <int KT, int NT, int ACOLS, int CCOLS>
__global__ __launch_bounds__(256) void gemm_wmma(
        const float* __restrict__ A, int lda,
        const _Float16* __restrict__ Bp,
        float* __restrict__ C, int ldc, int nrows) {
    constexpr int FRAGS = KT * NT;
    constexpr int CCPAD = ((CCOLS + 15) / 16) * 16;
    constexpr int Q     = CCOLS / 4;             // CCOLS is a multiple of 4

    __shared__ __align__(64) _Float16 sB[FRAGS * 512];
    __shared__ __align__(64) float    sC[8 * 16 * CCPAD];

    stage_B(Bp, sB, (unsigned)(FRAGS * 512 * sizeof(_Float16)));

    const int lane = threadIdx.x & 31;
    const int wave = threadIdx.x >> 5;
    const int row_base = (blockIdx.x * 8 + wave) * 16;
    if (row_base < nrows) {                      // wave-uniform; EXEC all-1 for WMMA
        const int lm = lane & 15;
        const int g  = (lane >> 4) & 1;

        v8f c[NT];
        v8f zero = {};
#pragma unroll
        for (int nt = 0; nt < NT; ++nt) c[nt] = zero;

        const float* Arow = A + (size_t)(row_base + lm) * lda;

#pragma unroll
        for (int kt = 0; kt < KT; ++kt) {
            const int kb = kt * 32 + g * 8;
            v4f p0 = *(const v4f*)(Arow + kb);
            v4f p1 = *(const v4f*)(Arow + kb + 4);
            v4f p2 = *(const v4f*)(Arow + kb + 16);
            v4f p3 = *(const v4f*)(Arow + kb + 20);
            v16h a;
#pragma unroll
            for (int j = 0; j < 8; ++j) {
                float v0 = (j < 4) ? p0[j] : p1[j - 4];
                float v1 = (j < 4) ? p2[j] : p3[j - 4];
                if (kb + j >= ACOLS)      v0 = 0.0f;   // folds when provable
                if (kb + 16 + j >= ACOLS) v1 = 0.0f;
                a[j]     = (_Float16)v0;
                a[8 + j] = (_Float16)v1;
            }
#pragma unroll
            for (int nt = 0; nt < NT; ++nt) {
                v16h b = *(const v16h*)(sB + ((kt * NT + nt) * 32 + lane) * 16);
                c[nt] = __builtin_amdgcn_wmma_f32_16x16x32_f16(
                    false, a, false, b, (short)0, c[nt], false, false);
            }
        }

        // transpose through per-wave LDS tile, then coalesced b128 stores
        float* myC = sC + wave * 16 * CCPAD;
#pragma unroll
        for (int nt = 0; nt < NT; ++nt)
#pragma unroll
            for (int r = 0; r < 8; ++r)
                myC[(g * 8 + r) * CCPAD + nt * 16 + lm] = c[nt][r];

        for (int idx = lane; idx < 16 * Q; idx += 32) {
            int row = idx / Q;
            int q   = idx - row * Q;
            int rr  = row_base + row;
            if (rr < nrows)
                *(v4f*)(C + (size_t)rr * ldc + q * 4) =
                    *(const v4f*)(myC + row * CCPAD + q * 4);
        }
    }
}

// ---------------------------------------------------------------------------
// Setup kernels
// ---------------------------------------------------------------------------
__global__ void init_nodes(const float* __restrict__ x, float* __restrict__ temp, int ldt,
                           float* __restrict__ ytm, float* __restrict__ deg, int n) {
    int i = blockIdx.x * blockDim.x + threadIdx.x;
    if (i >= n) return;
    float mean = 0.0f;
    for (int j = 0; j < TPAST; ++j) {
        float v = x[(size_t)i * TPAST + j];
        temp[(size_t)i * ldt + j] = v;
        mean += v;
    }
    ytm[i] = mean / (float)TPAST;
    deg[i] = 1.0f;                              // self-loop weight
}

__global__ void deg_edge(const int* __restrict__ dst, const float* __restrict__ ew,
                         float* __restrict__ deg, int E) {
    int e = blockIdx.x * blockDim.x + threadIdx.x;
    if (e >= E) return;
    atomicAdd(&deg[dst[e]], ew[e]);
}

__global__ void deg_fin(const float* __restrict__ deg, float* __restrict__ dinv, int n) {
    int i = blockIdx.x * blockDim.x + threadIdx.x;
    if (i >= n) return;
    float d = deg[i];
    dinv[i] = d > 0.0f ? rsqrtf(d) : 0.0f;
}

// ---------------------------------------------------------------------------
// GAT kernels (segment softmax over dst; self-loops handled in node kernels)
// ---------------------------------------------------------------------------
__global__ void gat_node_pre(const float* __restrict__ h, int ldh, int C,
                             const float* __restrict__ a_s, const float* __restrict__ a_d,
                             float* __restrict__ s, float* __restrict__ d,
                             float* __restrict__ lself, unsigned* __restrict__ mkey, int n) {
    int i = blockIdx.x * blockDim.x + threadIdx.x;
    if (i >= n) return;
    const float* hr = h + (size_t)i * ldh;
    float ss = 0.0f, dd = 0.0f;
    for (int c = 0; c < C; ++c) { float v = hr[c]; ss += v * a_s[c]; dd += v * a_d[c]; }
    s[i] = ss; d[i] = dd;
    float l0 = lrelu(ss + dd, 0.2f);            // self-loop logit
    lself[i] = l0;
    mkey[i]  = fkey(l0);                        // seeds segment max
}

__global__ void edge_max(const int* __restrict__ src, const int* __restrict__ dst,
                         const float* __restrict__ s, const float* __restrict__ d,
                         unsigned* __restrict__ mkey, int E) {
    int e = blockIdx.x * blockDim.x + threadIdx.x;
    if (e >= E) return;
    int u = src[e], v = dst[e];
    float l = lrelu(s[u] + d[v], 0.2f);
    atomicMax(&mkey[v], fkey(l));
}

__global__ void gat_node_init(const float* __restrict__ h, int ldh, int C,
                              const unsigned* __restrict__ mkey, const float* __restrict__ lself,
                              float* __restrict__ ssum, float* __restrict__ acc, int ldacc, int n) {
    int i = blockIdx.x * blockDim.x + threadIdx.x;
    if (i >= n) return;
    float m = funkey(mkey[i]);
    float ex = __expf(lself[i] - m);            // self-loop term
    ssum[i] = ex;
    const float* hr = h + (size_t)i * ldh;
    float* ar = acc + (size_t)i * ldacc;
    for (int c = 0; c < C; ++c) ar[c] = ex * hr[c];
}

__global__ void gat_edge_acc(const int* __restrict__ src, const int* __restrict__ dst,
                             const float* __restrict__ s, const float* __restrict__ d,
                             const unsigned* __restrict__ mkey,
                             const float* __restrict__ h, int ldh, int C,
                             float* __restrict__ ssum, float* __restrict__ acc, int ldacc, int E) {
    int e = blockIdx.x * blockDim.x + threadIdx.x;
    if (e >= E) return;
    int u = src[e], v = dst[e];
    float l = lrelu(s[u] + d[v], 0.2f);
    float ex = __expf(l - funkey(mkey[v]));
    atomicAdd(&ssum[v], ex);
    const float* hr = h + (size_t)u * ldh;
    float* ar = acc + (size_t)v * ldacc;
    for (int c = 0; c < C; ++c) atomicAdd(&ar[c], ex * hr[c]);
}

__global__ void gat_node_fin(const float* __restrict__ acc, int ldacc, int C,
                             const float* __restrict__ ssum, const float* __restrict__ bias,
                             float* __restrict__ out, int ldo, int ocol, float slope, int n) {
    int i = blockIdx.x * blockDim.x + threadIdx.x;
    if (i >= n) return;
    float inv = 1.0f / (ssum[i] + 1e-16f);
    const float* ar = acc + (size_t)i * ldacc;
    float* orow = out + (size_t)i * ldo + ocol;
    for (int c = 0; c < C; ++c) orow[c] = lrelu(ar[c] * inv + bias[c], slope);
}

// ---------------------------------------------------------------------------
// GCN kernels (symmetric norm; self-loop norm = dinv^2)
// ---------------------------------------------------------------------------
__global__ void gcn_node_init(const float* __restrict__ h, int ldh, int C,
                              const float* __restrict__ dinv,
                              float* __restrict__ acc, int ldacc, int n) {
    int i = blockIdx.x * blockDim.x + threadIdx.x;
    if (i >= n) return;
    float w = dinv[i] * dinv[i];
    const float* hr = h + (size_t)i * ldh;
    float* ar = acc + (size_t)i * ldacc;
    for (int c = 0; c < C; ++c) ar[c] = w * hr[c];
}

__global__ void gcn_edge_acc(const int* __restrict__ src, const int* __restrict__ dst,
                             const float* __restrict__ ew, const float* __restrict__ dinv,
                             const float* __restrict__ h, int ldh, int C,
                             float* __restrict__ acc, int ldacc, int E) {
    int e = blockIdx.x * blockDim.x + threadIdx.x;
    if (e >= E) return;
    int u = src[e], v = dst[e];
    float w = dinv[u] * ew[e] * dinv[v];
    const float* hr = h + (size_t)u * ldh;
    float* ar = acc + (size_t)v * ldacc;
    for (int c = 0; c < C; ++c) atomicAdd(&ar[c], w * hr[c]);
}

__global__ void gcn_node_fin(float* __restrict__ acc, int ldacc, int C,
                             const float* __restrict__ bias, float slope, int n) {
    int i = blockIdx.x * blockDim.x + threadIdx.x;
    if (i >= n) return;
    float* ar = acc + (size_t)i * ldacc;
    for (int c = 0; c < C; ++c) ar[c] = lrelu(ar[c] + bias[c], slope);   // in-place -> xt
}

// ---------------------------------------------------------------------------
// Output GAT (76 -> 1) + recurrent shift
// ---------------------------------------------------------------------------
__global__ void out_node_pre(const float* __restrict__ xt, int ldx,
                             const float* __restrict__ Wo,
                             const float* __restrict__ aso, const float* __restrict__ ado,
                             float* __restrict__ ho, float* __restrict__ s, float* __restrict__ d,
                             float* __restrict__ lself, unsigned* __restrict__ mkey, int n) {
    int i = blockIdx.x * blockDim.x + threadIdx.x;
    if (i >= n) return;
    const float* xr = xt + (size_t)i * ldx;
    float hv = 0.0f;
    for (int c = 0; c < TEMPC; ++c) hv += xr[c] * Wo[c];
    ho[i] = hv;
    float ss = hv * aso[0], dd = hv * ado[0];
    s[i] = ss; d[i] = dd;
    float l0 = lrelu(ss + dd, 0.2f);
    lself[i] = l0;
    mkey[i]  = fkey(l0);
}

__global__ void out_node_init(const unsigned* __restrict__ mkey, const float* __restrict__ lself,
                              const float* __restrict__ ho,
                              float* __restrict__ ssum, float* __restrict__ acc1, int n) {
    int i = blockIdx.x * blockDim.x + threadIdx.x;
    if (i >= n) return;
    float ex = __expf(lself[i] - funkey(mkey[i]));
    ssum[i] = ex;
    acc1[i] = ex * ho[i];
}

__global__ void out_edge_acc(const int* __restrict__ src, const int* __restrict__ dst,
                             const float* __restrict__ s, const float* __restrict__ d,
                             const unsigned* __restrict__ mkey, const float* __restrict__ ho,
                             float* __restrict__ ssum, float* __restrict__ acc1, int E) {
    int e = blockIdx.x * blockDim.x + threadIdx.x;
    if (e >= E) return;
    int u = src[e], v = dst[e];
    float l = lrelu(s[u] + d[v], 0.2f);
    float ex = __expf(l - funkey(mkey[v]));
    atomicAdd(&ssum[v], ex);
    atomicAdd(&acc1[v], ex * ho[u]);
}

__global__ void out_fin_shift(const float* __restrict__ acc1, const float* __restrict__ ssum,
                              const float* __restrict__ bo, float* __restrict__ ytm,
                              float* __restrict__ temp, int ldt,
                              float* __restrict__ out, int T, int t, int n) {
    int i = blockIdx.x * blockDim.x + threadIdx.x;
    if (i >= n) return;
    float yp = ytm[i] + acc1[i] / (ssum[i] + 1e-16f) + bo[0];
    out[(size_t)i * T + t] = yp;                 // unclipped; used for recurrence
    float* xr = temp + (size_t)i * ldt;
    float mean = 0.0f;
    for (int j = 0; j < TPAST - 1; ++j) { float v = xr[j + 1]; xr[j] = v; mean += v; }
    xr[TPAST - 1] = yp;
    mean += yp;
    ytm[i] = mean / (float)TPAST;
}

__global__ void clip_k(float* __restrict__ out, int tot) {
    int i = blockIdx.x * blockDim.x + threadIdx.x;
    if (i >= tot) return;
    out[i] = fminf(fmaxf(out[i], 0.0f), 90.0f);
}

// ---------------------------------------------------------------------------
extern "C" void kernel_launch(void* const* d_in, const int* in_sizes, int n_in,
                              void* d_out, int out_size, void* d_ws, size_t ws_size,
                              hipStream_t stream) {
    const float* x   = (const float*)d_in[0];
    const int*   ei  = (const int*)d_in[1];
    const float* ew  = (const float*)d_in[2];
    const float* W1  = (const float*)d_in[4];
    const float* as1 = (const float*)d_in[5];
    const float* ad1 = (const float*)d_in[6];
    const float* b1  = (const float*)d_in[7];
    const float* W2  = (const float*)d_in[8];
    const float* as2 = (const float*)d_in[9];
    const float* ad2 = (const float*)d_in[10];
    const float* b2  = (const float*)d_in[11];
    const float* Wg  = (const float*)d_in[12];
    const float* bg  = (const float*)d_in[13];
    const float* Wo  = (const float*)d_in[14];
    const float* aso = (const float*)d_in[15];
    const float* ado = (const float*)d_in[16];
    const float* bo  = (const float*)d_in[17];

    const int N = in_sizes[0] / TPAST;
    const int E = in_sizes[1] / 2;
    const int T = out_size / N;                  // T_future (no device read needed)
    const int* src = ei;
    const int* dst = ei + E;
    float* out = (float*)d_out;

    // carve workspace (256B aligned)
    char* w = (char*)d_ws;
    auto carve = [&](size_t bytes) -> void* {
        void* p = (void*)w;
        w += (bytes + 255) & ~(size_t)255;
        return p;
    };
    float*    temp  = (float*)carve((size_t)N * TEMPC * 4);   // [x | h1 | h2]
    float*    h     = (float*)carve((size_t)N * TEMPC * 4);   // pre-aggregation features
    float*    acc   = (float*)carve((size_t)N * TEMPC * 4);   // aggregation / xt (in place)
    float*    s     = (float*)carve((size_t)N * 4);
    float*    d     = (float*)carve((size_t)N * 4);
    float*    lself = (float*)carve((size_t)N * 4);
    float*    ssum  = (float*)carve((size_t)N * 4);
    unsigned* mkey  = (unsigned*)carve((size_t)N * 4);
    float*    deg   = (float*)carve((size_t)N * 4);
    float*    dinv  = (float*)carve((size_t)N * 4);
    float*    ytm   = (float*)carve((size_t)N * 4);
    float*    ho    = (float*)carve((size_t)N * 4);
    float*    acc1  = (float*)carve((size_t)N * 4);
    _Float16* Bp1   = (_Float16*)carve((size_t)1 * 2 * 32 * 16 * 2);   // 12x32  -> KT=1,NT=2
    _Float16* Bp2   = (_Float16*)carve((size_t)1 * 2 * 32 * 16 * 2);   // 32x32  -> KT=1,NT=2
    _Float16* Bp3   = (_Float16*)carve((size_t)3 * 5 * 32 * 16 * 2);   // 76x76  -> KT=3,NT=5

    const int TB = 256;
    dim3 nb((N + TB - 1) / TB), ebk((E + TB - 1) / TB), tb(TB);
    dim3 gbk((N + 127) / 128);                   // 8 waves/block * 16 rows/wave

    // ---- one-time setup ----
    init_nodes<<<nb, tb, 0, stream>>>(x, temp, TEMPC, ytm, deg, N);
    deg_edge<<<ebk, tb, 0, stream>>>(dst, ew, deg, E);
    deg_fin<<<nb, tb, 0, stream>>>(deg, dinv, N);
    pack_b<<<1, 64, 0, stream>>>(W1, Bp1, TPAST, HIDC, 1, 2);
    pack_b<<<1, 64, 0, stream>>>(W2, Bp2, HIDC, HIDC, 1, 2);
    pack_b<<<(3 * 5 * 32 + 63) / 64, 64, 0, stream>>>(Wg, Bp3, TEMPC, TEMPC, 3, 5);

    for (int t = 0; t < T; ++t) {
        // ---- GAT1: x_init (temp cols 0..11) -> h1 (temp cols 12..43) ----
        gemm_wmma<1, 2, TPAST, HIDC><<<gbk, tb, 0, stream>>>(temp, TEMPC, Bp1, h, TEMPC, N);
        gat_node_pre<<<nb, tb, 0, stream>>>(h, TEMPC, HIDC, as1, ad1, s, d, lself, mkey, N);
        edge_max<<<ebk, tb, 0, stream>>>(src, dst, s, d, mkey, E);
        gat_node_init<<<nb, tb, 0, stream>>>(h, TEMPC, HIDC, mkey, lself, ssum, acc, TEMPC, N);
        gat_edge_acc<<<ebk, tb, 0, stream>>>(src, dst, s, d, mkey, h, TEMPC, HIDC, ssum, acc, TEMPC, E);
        gat_node_fin<<<nb, tb, 0, stream>>>(acc, TEMPC, HIDC, ssum, b1, temp, TEMPC, TPAST, 0.01f, N);

        // ---- GAT2: h1 -> h2 (temp cols 44..75) ----
        gemm_wmma<1, 2, HIDC, HIDC><<<gbk, tb, 0, stream>>>(temp + TPAST, TEMPC, Bp2, h, TEMPC, N);
        gat_node_pre<<<nb, tb, 0, stream>>>(h, TEMPC, HIDC, as2, ad2, s, d, lself, mkey, N);
        edge_max<<<ebk, tb, 0, stream>>>(src, dst, s, d, mkey, E);
        gat_node_init<<<nb, tb, 0, stream>>>(h, TEMPC, HIDC, mkey, lself, ssum, acc, TEMPC, N);
        gat_edge_acc<<<ebk, tb, 0, stream>>>(src, dst, s, d, mkey, h, TEMPC, HIDC, ssum, acc, TEMPC, E);
        gat_node_fin<<<nb, tb, 0, stream>>>(acc, TEMPC, HIDC, ssum, b2, temp, TEMPC, TPAST + HIDC, 0.01f, N);

        // ---- GCN: temp (76) -> xt (in acc, in place) ----
        gemm_wmma<3, 5, TEMPC, TEMPC><<<gbk, tb, 0, stream>>>(temp, TEMPC, Bp3, h, TEMPC, N);
        gcn_node_init<<<nb, tb, 0, stream>>>(h, TEMPC, TEMPC, dinv, acc, TEMPC, N);
        gcn_edge_acc<<<ebk, tb, 0, stream>>>(src, dst, ew, dinv, h, TEMPC, TEMPC, acc, TEMPC, E);
        gcn_node_fin<<<nb, tb, 0, stream>>>(acc, TEMPC, TEMPC, bg, 0.01f, N);

        // ---- output GAT (76 -> 1), prediction, recurrent shift ----
        out_node_pre<<<nb, tb, 0, stream>>>(acc, TEMPC, Wo, aso, ado, ho, s, d, lself, mkey, N);
        edge_max<<<ebk, tb, 0, stream>>>(src, dst, s, d, mkey, E);
        out_node_init<<<nb, tb, 0, stream>>>(mkey, lself, ho, ssum, acc1, N);
        out_edge_acc<<<ebk, tb, 0, stream>>>(src, dst, s, d, mkey, ho, ssum, acc1, E);
        out_fin_shift<<<nb, tb, 0, stream>>>(acc1, ssum, bo, ytm, temp, TEMPC, out, T, t, N);
    }

    clip_k<<<(N * T + TB - 1) / TB, tb, 0, stream>>>(out, N * T);
}
